// HierarchicalAnchorNet_29248727286400
// MI455X (gfx1250) — matrix operand order repaired and checked by reading, not compiled, not run on hardware
//
#include <hip/hip_runtime.h>
#include <cstdint>
#include <cstddef>

typedef __attribute__((ext_vector_type(16))) _Float16 v16h;
typedef __attribute__((ext_vector_type(8)))  float    v8f;

namespace {
constexpr int NPTS  = 131072;   // N = B*P
constexpr int PPB   = 8192;     // P
constexpr int ATOT  = 2048;     // A = B*NA
constexpr int NEDGE = 65536;    // E
constexpr int NPART = 128;      // partial blocks for BN reductions
}

// =====================  BN: deterministic 2-stage stats  =====================
__global__ void k_bn_partial(const float* __restrict__ X, int ldx, int cox,
                             int M, int C, float* __restrict__ part)
{
  int c = threadIdx.x;
  if (c >= C) return;
  float s = 0.f, ss = 0.f;
  for (int r = blockIdx.x; r < M; r += gridDim.x) {
    float v = X[(size_t)r * ldx + cox + c];
    s += v; ss += v * v;
  }
  part[(size_t)blockIdx.x * (2 * C) + c]     = s;
  part[(size_t)blockIdx.x * (2 * C) + C + c] = ss;
}

__global__ void k_bn_final(const float* __restrict__ part, int npart, int C, float invM,
                           const float* __restrict__ g, const float* __restrict__ b,
                           float* __restrict__ sc, float* __restrict__ sh)
{
  int c = blockIdx.x * blockDim.x + threadIdx.x;
  if (c >= C) return;
  float s = 0.f, ss = 0.f;
  for (int p = 0; p < npart; ++p) {
    s  += part[(size_t)p * (2 * C) + c];
    ss += part[(size_t)p * (2 * C) + C + c];
  }
  float mu    = s * invM;
  float var   = ss * invM - mu * mu;           // biased var, matches jnp .var(0)
  float scale = g[c] * rsqrtf(var + 1e-5f);
  sc[c] = scale;
  sh[c] = b[c] - mu * scale;
}

__global__ void k_bn_apply16(const float* __restrict__ X, int ldx, int cox,
                             _Float16* __restrict__ Y, int ldy, int coy,
                             const float* __restrict__ sc, const float* __restrict__ sh,
                             int M, int C, int relu)
{
  size_t tot = (size_t)M * C;
  for (size_t i = (size_t)blockIdx.x * blockDim.x + threadIdx.x; i < tot;
       i += (size_t)gridDim.x * blockDim.x) {
    int r = (int)(i / C), c = (int)(i % C);
    float v = X[(size_t)r * ldx + cox + c] * sc[c] + sh[c];
    if (relu) v = fmaxf(v, 0.f);
    Y[(size_t)r * ldy + coy + c] = (_Float16)v;
  }
}

__global__ void k_bn_apply32(const float* __restrict__ X, int ldx, int cox,
                             float* __restrict__ Y, int ldy, int coy,
                             const float* __restrict__ sc, const float* __restrict__ sh,
                             int M, int C, int relu)
{
  size_t tot = (size_t)M * C;
  for (size_t i = (size_t)blockIdx.x * blockDim.x + threadIdx.x; i < tot;
       i += (size_t)gridDim.x * blockDim.x) {
    int r = (int)(i / C), c = (int)(i % C);
    float v = X[(size_t)r * ldx + cox + c] * sc[c] + sh[c];
    if (relu) v = fmaxf(v, 0.f);
    Y[(size_t)r * ldy + coy + c] = v;
  }
}

__global__ void k_cvt16(const float* __restrict__ X, int ldx, int cox,
                        _Float16* __restrict__ Y, int ldy, int coy, int M, int C, int relu)
{
  size_t tot = (size_t)M * C;
  for (size_t i = (size_t)blockIdx.x * blockDim.x + threadIdx.x; i < tot;
       i += (size_t)gridDim.x * blockDim.x) {
    int r = (int)(i / C), c = (int)(i % C);
    float v = X[(size_t)r * ldx + cox + c];
    if (relu) v = fmaxf(v, 0.f);
    Y[(size_t)r * ldy + coy + c] = (_Float16)v;
  }
}

// =============  weight prep: fp32 (K x N) -> f16 transposed (N x Kpad)  ======
__global__ void k_prep_w(const float* __restrict__ W, int ldw, int krow0, int Kc, int Nout,
                         _Float16* __restrict__ Wt, int ldk, int kofs)
{
  size_t tot = (size_t)Kc * Nout;
  for (size_t i = (size_t)blockIdx.x * blockDim.x + threadIdx.x; i < tot;
       i += (size_t)gridDim.x * blockDim.x) {
    int n = (int)(i / Kc), k = (int)(i % Kc);
    Wt[(size_t)n * ldk + kofs + k] = (_Float16)W[(size_t)(krow0 + k) * ldw + n];
  }
}

// =====================  WMMA GEMM: C = act(A*W^T + bias)  ====================
// A: M x lda f16 (K zero-padded).  Wt: Npad x ldk f16 (pre-transposed, zero-padded).
// One wave -> FOUR 16x16 M-tiles x one N-tile (64x16 of C): the B fragment is
// loaded once per K-step and reused by 4 v_wmma_f32_16x16x32_f16, so per K-step
// a wave issues 10 b128 loads for 4 WMMAs (vs 16 for 4 independent tiles).
// M is a multiple of 64 for every call site.
// A-frag: lane (half=l>>4, lr=l&15) reads row (tile_m*16+lr), two 16B chunks at
//   k0+half*8 and k0+half*8+16  (ISA 16-bit A 16x32 layout).
// B-frag: lane reads Wt row (tn*16+lr), two 16B chunks at k0+half*8 and k0+half*8+16.
__global__ void k_gemm(const _Float16* __restrict__ Aact, int lda,
                       const _Float16* __restrict__ Wt, int ldk,
                       const float* __restrict__ bias,
                       float* __restrict__ Cout, int ldc,
                       int M, int Nv, int K, int relu)
{
  int wave = threadIdx.x >> 5;
  int lane = threadIdx.x & 31;
  int tiles_n = (Nv + 15) >> 4;
  int mgrp = M >> 6;                  // groups of 4 M-tiles
  int tile = blockIdx.x * 4 + wave;   // 4 waves per block; tn varies fastest
  if (tile >= mgrp * tiles_n) return;
  int tg = tile / tiles_n, tn = tile % tiles_n;
  int half_ = lane >> 4, lr = lane & 15;

  const _Float16* arow = Aact + (size_t)(tg * 64 + lr) * lda + half_ * 8;
  const _Float16* brow = Wt   + (size_t)(tn * 16 + lr) * ldk + half_ * 8;
  const size_t astep = (size_t)16 * lda;

  v8f acc0 = {}, acc1 = {}, acc2 = {}, acc3 = {};
  for (int k0 = 0; k0 < K; k0 += 32) {
    v16h b;
    ((uint4*)&b)[0] = *(const uint4*)(brow + k0);
    ((uint4*)&b)[1] = *(const uint4*)(brow + k0 + 16);
    v16h a0, a1, a2, a3;
    const _Float16* p = arow + k0;
    ((uint4*)&a0)[0] = *(const uint4*)(p);
    ((uint4*)&a0)[1] = *(const uint4*)(p + 16);
    p += astep;
    ((uint4*)&a1)[0] = *(const uint4*)(p);
    ((uint4*)&a1)[1] = *(const uint4*)(p + 16);
    p += astep;
    ((uint4*)&a2)[0] = *(const uint4*)(p);
    ((uint4*)&a2)[1] = *(const uint4*)(p + 16);
    p += astep;
    ((uint4*)&a3)[0] = *(const uint4*)(p);
    ((uint4*)&a3)[1] = *(const uint4*)(p + 16);
    acc0 = __builtin_amdgcn_wmma_f32_16x16x32_f16(false, a0, false, b, (short)0, acc0, false, false);
    acc1 = __builtin_amdgcn_wmma_f32_16x16x32_f16(false, a1, false, b, (short)0, acc1, false, false);
    acc2 = __builtin_amdgcn_wmma_f32_16x16x32_f16(false, a2, false, b, (short)0, acc2, false, false);
    acc3 = __builtin_amdgcn_wmma_f32_16x16x32_f16(false, a3, false, b, (short)0, acc3, false, false);
  }

  int n = tn * 16 + lr;
  if (n < Nv) {
    float bv = bias[n];
    v8f accs[4] = {acc0, acc1, acc2, acc3};
    #pragma unroll
    for (int t = 0; t < 4; ++t) {
      int mbase = tg * 64 + t * 16 + half_ * 8;
      #pragma unroll
      for (int i = 0; i < 8; ++i) {
        float v = accs[t][i] + bv;
        if (relu) v = fmaxf(v, 0.f);
        Cout[(size_t)(mbase + i) * ldc + n] = v;
      }
    }
  }
}

// =====================  edge feature build  ==================================
__global__ void k_edge_build(const float* __restrict__ xsrc, int cin,
                             const int* __restrict__ row, const int* __restrict__ col,
                             _Float16* __restrict__ eout /* E x 32 */)
{
  int e = blockIdx.x * blockDim.x + threadIdx.x;
  if (e >= NEDGE) return;
  int r = row[e], c = col[e];
  for (int j = 0; j < cin; ++j) {
    float xr = xsrc[(size_t)r * cin + j];
    float xc = xsrc[(size_t)c * cin + j];
    eout[(size_t)e * 32 + j]       = (_Float16)xr;
    eout[(size_t)e * 32 + cin + j] = (_Float16)(xr - xc);
  }
}

// =====  segment max + segment softmax-attention (32 contiguous edges/anchor) =
__global__ void k_segment(const _Float16* __restrict__ ef /* E x128 */,
                          const float* __restrict__ logit /* E x16, col0 */,
                          _Float16* __restrict__ out /* A x 256: [max|pad|att|pad] */)
{
  __shared__ float alpha[32];
  int a = blockIdx.x, t = threadIdx.x;
  if (t == 0) {
    float m = -1e30f;
    for (int j = 0; j < 32; ++j) m = fmaxf(m, logit[(size_t)(a * 32 + j) * 16]);
    float ex[32]; float z = 0.f;
    for (int j = 0; j < 32; ++j) { ex[j] = __expf(logit[(size_t)(a * 32 + j) * 16] - m); z += ex[j]; }
    float inv = 1.f / (z + 1e-16f);
    for (int j = 0; j < 32; ++j) alpha[j] = ex[j] * inv;
  }
  __syncthreads();
  for (int c = t; c < 126; c += blockDim.x) {
    float mx = -1e30f, s = 0.f;
    for (int j = 0; j < 32; ++j) {
      float v = (float)ef[(size_t)(a * 32 + j) * 128 + c];
      mx = fmaxf(mx, v);
      s += v * alpha[j];
    }
    if (!__builtin_isfinite(mx)) mx = 0.f;
    out[(size_t)a * 256 + c]       = (_Float16)mx;
    out[(size_t)a * 256 + 128 + c] = (_Float16)s;
  }
}

// =====================  anchor pos gather  ===================================
__global__ void k_gather_pos(const float* __restrict__ x, const int* __restrict__ aidx,
                             float* __restrict__ apos)
{
  int a = blockIdx.x * blockDim.x + threadIdx.x;
  if (a >= ATOT) return;
  int p = aidx[a];
  apos[a * 3 + 0] = x[(size_t)p * 8 + 0];
  apos[a * 3 + 1] = x[(size_t)p * 8 + 1];
  apos[a * 3 + 2] = x[(size_t)p * 8 + 2];
}

// =====================  attention (per b,h block; 128 tokens, 64 dim) ========
__global__ void k_attention(const float* __restrict__ qkv /* A x 768 */,
                            float* __restrict__ o /* A x 256 */)
{
  __shared__ _Float16 sK[128 * 64];
  __shared__ _Float16 sV[128 * 64];
  int bh = blockIdx.x, b = bh >> 2, h = bh & 3;
  int t = threadIdx.x; // 0..127 : one query token
  const float* base = qkv + (size_t)b * 128 * 768;
  for (int j = 0; j < 64; ++j) {
    sK[t * 64 + j] = (_Float16)base[(size_t)t * 768 + 256 + h * 64 + j];
    sV[t * 64 + j] = (_Float16)base[(size_t)t * 768 + 512 + h * 64 + j];
  }
  __syncthreads();
  float q[64];
  for (int j = 0; j < 64; ++j) q[j] = base[(size_t)t * 768 + h * 64 + j];
  const float scale = 0.125f; // 1/sqrt(64)
  float m = -1e30f;
  for (int j = 0; j < 128; ++j) {
    float d = 0.f;
    for (int l = 0; l < 64; ++l) d += q[l] * (float)sK[j * 64 + l];
    m = fmaxf(m, d * scale);
  }
  float z = 0.f, accv[64];
  for (int l = 0; l < 64; ++l) accv[l] = 0.f;
  for (int j = 0; j < 128; ++j) {
    float d = 0.f;
    for (int l = 0; l < 64; ++l) d += q[l] * (float)sK[j * 64 + l];
    float p = __expf(d * scale - m);
    z += p;
    for (int l = 0; l < 64; ++l) accv[l] += p * (float)sV[j * 64 + l];
  }
  float invz = 1.f / z;
  float* op = o + (size_t)(b * 128 + t) * 256 + h * 64;
  for (int l = 0; l < 64; ++l) op[l] = accv[l] * invz;
}

// =====================  layernorm over D=256 with residual  ==================
__global__ void k_layernorm(const float* __restrict__ X, const float* __restrict__ R,
                            const float* __restrict__ g, const float* __restrict__ bb,
                            float* __restrict__ Y)
{
  __shared__ float red[256];
  int r = blockIdx.x, t = threadIdx.x;
  float v = X[(size_t)r * 256 + t] + R[(size_t)r * 256 + t];
  red[t] = v; __syncthreads();
  for (int s = 128; s > 0; s >>= 1) { if (t < s) red[t] += red[t + s]; __syncthreads(); }
  float mu = red[0] * (1.f / 256.f); __syncthreads();
  float d = v - mu;
  red[t] = d * d; __syncthreads();
  for (int s = 128; s > 0; s >>= 1) { if (t < s) red[t] += red[t + s]; __syncthreads(); }
  float var = red[0] * (1.f / 256.f);
  Y[(size_t)r * 256 + t] = d * rsqrtf(var + 1e-5f) * g[t] + bb[t];
}

// ============  3-NN inverse-distance interp + classifier input build  ========
__global__ void k_knn(const float* __restrict__ x, const float* __restrict__ apos,
                      const float* __restrict__ ag /* A x 256 */,
                      const float* __restrict__ xn,
                      _Float16* __restrict__ cin /* N x 288 */)
{
  __shared__ float d2s[128];
  __shared__ float w[3]; __shared__ int id[3]; __shared__ float wsum;
  int p = blockIdx.x, b = p >> 13; // /8192
  int t = threadIdx.x;             // 128 threads: one anchor each
  float px = x[(size_t)p * 8 + 0], py = x[(size_t)p * 8 + 1], pz = x[(size_t)p * 8 + 2];
  const float* ap = apos + (size_t)b * 128 * 3;
  {
    float dx = px - ap[t * 3], dy = py - ap[t * 3 + 1], dz = pz - ap[t * 3 + 2];
    d2s[t] = dx * dx + dy * dy + dz * dz;
  }
  __syncthreads();
  if (t == 0) {
    int i0 = 0, i1 = 0, i2 = 0; float b0 = 1e30f, b1 = 1e30f, b2 = 1e30f;
    for (int j = 0; j < 128; ++j) {
      float d = d2s[j];
      if (d < b0)      { b2 = b1; i2 = i1; b1 = b0; i1 = i0; b0 = d; i0 = j; }
      else if (d < b1) { b2 = b1; i2 = i1; b1 = d;  i1 = j; }
      else if (d < b2) { b2 = d;  i2 = j; }
    }
    float w0 = 1.f / fmaxf(b0, 1e-16f), w1 = 1.f / fmaxf(b1, 1e-16f), w2 = 1.f / fmaxf(b2, 1e-16f);
    w[0] = w0; w[1] = w1; w[2] = w2; id[0] = i0; id[1] = i1; id[2] = i2; wsum = w0 + w1 + w2;
  }
  __syncthreads();
  const float* a0 = ag + (size_t)(b * 128 + id[0]) * 256;
  const float* a1 = ag + (size_t)(b * 128 + id[1]) * 256;
  const float* a2 = ag + (size_t)(b * 128 + id[2]) * 256;
  float invs = 1.f / wsum;
  _Float16* row = cin + (size_t)p * 288;
  for (int d = t; d < 256; d += 128)
    row[d] = (_Float16)((w[0] * a0[d] + w[1] * a1[d] + w[2] * a2[d]) * invs);
  if (t < 8) row[256 + t] = (_Float16)xn[(size_t)p * 8 + t];
}

// ============================  host driver  ==================================
extern "C" void kernel_launch(void* const* d_in, const int* in_sizes, int n_in,
                              void* d_out, int out_size, void* d_ws, size_t ws_size,
                              hipStream_t stream)
{
  (void)in_sizes; (void)n_in; (void)out_size; (void)ws_size;
  const float* x  = (const float*)d_in[0];
  const int* aidx = (const int*)d_in[1];
  const int* erow = (const int*)d_in[2];
  const int* ecol = (const int*)d_in[3];
  auto F = [&](int i) { return (const float*)d_in[i]; };
  // param flat order (depth-first, insertion order):
  // 4,5 in_bn g/b | 6,7 pos_bn | spa: 8..23 | tmp: 24..39 | 40,41 stream w/b
  // 42,43 bn_fuse | 44,45 attn_in | 46,47 attn_out | 48,49 ln1 | 50,51 ln2
  // 52,53 ff1 | 54,55 ff2 | 56,57 cls1 | 58,59 cbn1 | 60,61 cls2 | 62,63 cbn2 | 64,65 cls3

  char* base = (char*)d_ws;
  size_t off = 0;
  auto alloc = [&](size_t bytes) -> char* {
    char* p = base + off;
    off = (off + bytes + 255) & ~(size_t)255;
    return p;
  };

  float*    xn    = (float*)alloc((size_t)NPTS * 8 * 4);
  float*    xs    = (float*)alloc((size_t)NPTS * 3 * 4);
  float*    xt    = (float*)alloc((size_t)NPTS * 5 * 4);
  float*    apos  = (float*)alloc((size_t)ATOT * 3 * 4);
  _Float16* seg[2]; seg[0] = (_Float16*)alloc((size_t)ATOT * 256 * 2);
  seg[1] = (_Float16*)alloc((size_t)ATOT * 256 * 2);
  _Float16* strin = (_Float16*)alloc((size_t)ATOT * 256 * 2);
  float*    fuo   = (float*)alloc((size_t)ATOT * 126 * 4);
  float*    stro  = (float*)alloc((size_t)ATOT * 253 * 4);
  float*    gi32  = (float*)alloc((size_t)ATOT * 256 * 4);
  _Float16* gi16  = (_Float16*)alloc((size_t)ATOT * 256 * 2);
  float*    qkv   = (float*)alloc((size_t)ATOT * 768 * 4);
  float*    o32   = (float*)alloc((size_t)ATOT * 256 * 4);
  _Float16* o16   = (_Float16*)alloc((size_t)ATOT * 256 * 2);
  float*    tA32  = (float*)alloc((size_t)ATOT * 256 * 4);
  float*    h132  = (float*)alloc((size_t)ATOT * 256 * 4);
  _Float16* h116  = (_Float16*)alloc((size_t)ATOT * 256 * 2);
  float*    ff132 = (float*)alloc((size_t)ATOT * 506 * 4);
  _Float16* ff116 = (_Float16*)alloc((size_t)ATOT * 512 * 2);
  float*    ag32  = (float*)alloc((size_t)ATOT * 256 * 4);
  float*    part  = (float*)alloc((size_t)NPART * 1024 * 4);
  float*    sc    = (float*)alloc(512 * 4);
  float*    sh    = (float*)alloc(512 * 4);

  // f16 transposed weights (zero padded)
  char* wreg0 = base + off;
  _Float16* wl1[2];  wl1[0] = (_Float16*)alloc(128 * 32 * 2);  wl1[1] = (_Float16*)alloc(128 * 32 * 2);
  _Float16* wl2[2];  wl2[0] = (_Float16*)alloc(128 * 128 * 2); wl2[1] = (_Float16*)alloc(128 * 128 * 2);
  _Float16* watt[2]; watt[0] = (_Float16*)alloc(16 * 128 * 2); watt[1] = (_Float16*)alloc(16 * 128 * 2);
  _Float16* wfu[2];  wfu[0] = (_Float16*)alloc(128 * 256 * 2); wfu[1] = (_Float16*)alloc(128 * 256 * 2);
  _Float16* wstr = (_Float16*)alloc(256 * 256 * 2);
  _Float16* wai  = (_Float16*)alloc(768 * 256 * 2);
  _Float16* wao  = (_Float16*)alloc(256 * 256 * 2);
  _Float16* wf1  = (_Float16*)alloc(512 * 256 * 2);
  _Float16* wf2  = (_Float16*)alloc(256 * 512 * 2);
  _Float16* wc1  = (_Float16*)alloc(64 * 288 * 2);
  _Float16* wc2  = (_Float16*)alloc(32 * 64 * 2);
  _Float16* wc3  = (_Float16*)alloc(16 * 32 * 2);
  size_t wreg_bytes = (size_t)((base + off) - wreg0);

  // big region: edge phase reused by classifier phase
  char* big = alloc(150994944);
  _Float16* e_in  = (_Float16*)(big);
  float*    g132  = (float*)(big + 4194304);
  _Float16* g116  = (_Float16*)(big + 37224448);
  float*    g232  = (float*)(big + 54001664);
  _Float16* efeat = (_Float16*)(big + 87031808);
  float*    logit = (float*)(big + 103809024);
  _Float16* clsin = (_Float16*)(big);
  float*    c132  = (float*)(big + 75497472);
  _Float16* c116  = (_Float16*)(big + 109051904);
  float*    c232  = (float*)(big + 125829120);
  _Float16* c216  = (_Float16*)(big + 142606336);

  auto gemm = [&](const _Float16* A, int lda, const _Float16* Wt, int ldk,
                  const float* bias, float* C, int ldc, int M, int Nv, int K, int relu) {
    int tiles = (M >> 6) * ((Nv + 15) >> 4);   // 64-row groups x 16-col tiles
    k_gemm<<<dim3((tiles + 3) / 4), dim3(128), 0, stream>>>(A, lda, Wt, ldk, bias, C, ldc, M, Nv, K, relu);
  };
  auto bn_stats = [&](const float* X, int ldx, int cox, int M, int C,
                      const float* g, const float* b) {
    k_bn_partial<<<dim3(NPART), dim3(256), 0, stream>>>(X, ldx, cox, M, C, part);
    k_bn_final<<<dim3((C + 255) / 256), dim3(256), 0, stream>>>(part, NPART, C, 1.f / (float)M, g, b, sc, sh);
  };
  auto prep = [&](const float* W, int ldw, int krow0, int Kc, int Nout,
                  _Float16* Wt, int ldk, int kofs) {
    k_prep_w<<<dim3(256), dim3(256), 0, stream>>>(W, ldw, krow0, Kc, Nout, Wt, ldk, kofs);
  };

  // ---- zero pads ----
  hipMemsetAsync(wreg0, 0, wreg_bytes, stream);
  hipMemsetAsync(e_in, 0, (size_t)NEDGE * 32 * 2, stream);
  hipMemsetAsync(g116, 0, (size_t)NEDGE * 128 * 2, stream);
  hipMemsetAsync(efeat, 0, (size_t)NEDGE * 128 * 2, stream);
  hipMemsetAsync(seg[0], 0, (size_t)ATOT * 256 * 2, stream);
  hipMemsetAsync(seg[1], 0, (size_t)ATOT * 256 * 2, stream);
  hipMemsetAsync(strin, 0, (size_t)ATOT * 256 * 2, stream);
  hipMemsetAsync(ff116, 0, (size_t)ATOT * 512 * 2, stream);

  // ---- weight prep ----
  prep(F(10), 126, 0, 6, 126, wl1[0], 32, 0);           // spa.l1
  prep(F(26), 126, 0, 10, 126, wl1[1], 32, 0);          // tmp.l1
  prep(F(14), 126, 0, 126, 126, wl2[0], 128, 0);        // spa.l2
  prep(F(30), 126, 0, 126, 126, wl2[1], 128, 0);        // tmp.l2
  prep(F(18), 1, 0, 126, 1, watt[0], 128, 0);           // spa.att
  prep(F(34), 1, 0, 126, 1, watt[1], 128, 0);           // tmp.att
  prep(F(20), 126, 0, 126, 126, wfu[0], 256, 0);        // spa.fuse (max|att split)
  prep(F(20), 126, 126, 126, 126, wfu[0], 256, 128);
  prep(F(36), 126, 0, 126, 126, wfu[1], 256, 0);        // tmp.fuse
  prep(F(36), 126, 126, 126, 126, wfu[1], 256, 128);
  prep(F(40), 253, 0, 126, 253, wstr, 256, 0);          // stream (spa|tmp split)
  prep(F(40), 253, 126, 126, 253, wstr, 256, 128);
  prep(F(44), 768, 0, 256, 768, wai, 256, 0);           // attn_in
  prep(F(46), 256, 0, 256, 256, wao, 256, 0);           // attn_out
  prep(F(52), 506, 0, 256, 506, wf1, 256, 0);           // ff1
  prep(F(54), 256, 0, 506, 256, wf2, 512, 0);           // ff2 (K 506 -> pad 512)
  prep(F(56), 64, 0, 256, 64, wc1, 288, 0);             // cls1 (interp part)
  prep(F(56), 64, 256, 8, 64, wc1, 288, 256);           // cls1 (xn part)
  prep(F(60), 32, 0, 64, 32, wc2, 64, 0);               // cls2
  prep(F(64), 1, 0, 32, 1, wc3, 32, 0);                 // cls3

  // ---- input BN: x -> xn, then per-stream input BN -> xs / xt ----
  bn_stats(x, 8, 0, NPTS, 8, F(4), F(5));
  k_bn_apply32<<<dim3(2048), dim3(256), 0, stream>>>(x, 8, 0, xn, 8, 0, sc, sh, NPTS, 8, 0);
  bn_stats(xn, 8, 0, NPTS, 3, F(8), F(9));
  k_bn_apply32<<<dim3(2048), dim3(256), 0, stream>>>(xn, 8, 0, xs, 3, 0, sc, sh, NPTS, 3, 0);
  bn_stats(xn, 8, 3, NPTS, 5, F(24), F(25));
  k_bn_apply32<<<dim3(2048), dim3(256), 0, stream>>>(xn, 8, 3, xt, 5, 0, sc, sh, NPTS, 5, 0);

  // ---- two aggregate streams (spa: cin=3, tmp: cin=5) ----
  for (int s = 0; s < 2; ++s) {
    const float* xsrc = (s == 0) ? xs : xt;
    int cin = (s == 0) ? 3 : 5;
    int pb = (s == 0) ? 10 : 26;  // l1.w base index
    k_edge_build<<<dim3(NEDGE / 256), dim3(256), 0, stream>>>(xsrc, cin, erow, ecol, e_in);
    gemm(e_in, 32, wl1[s], 32, F(pb + 1), g132, 126, NEDGE, 126, 32, 1);          // l1 + relu
    bn_stats(g132, 126, 0, NEDGE, 126, F(pb + 2), F(pb + 3));                     // bn1
    k_bn_apply16<<<dim3(4096), dim3(256), 0, stream>>>(g132, 126, 0, g116, 128, 0, sc, sh, NEDGE, 126, 0);
    gemm(g116, 128, wl2[s], 128, F(pb + 5), g232, 126, NEDGE, 126, 128, 1);       // l2 + relu
    bn_stats(g232, 126, 0, NEDGE, 126, F(pb + 6), F(pb + 7));                     // bn2
    k_bn_apply16<<<dim3(4096), dim3(256), 0, stream>>>(g232, 126, 0, efeat, 128, 0, sc, sh, NEDGE, 126, 0);
    gemm(efeat, 128, watt[s], 128, F(pb + 9), logit, 16, NEDGE, 1, 128, 0);       // att logits
    k_segment<<<dim3(ATOT), dim3(128), 0, stream>>>(efeat, logit, seg[s]);
    gemm(seg[s], 256, wfu[s], 256, F(pb + 11), fuo, 126, ATOT, 126, 256, 0);      // fuse
    bn_stats(fuo, 126, 0, ATOT, 126, F(pb + 12), F(pb + 13));                     // out_bn
    k_bn_apply16<<<dim3(512), dim3(256), 0, stream>>>(fuo, 126, 0, strin, 256, s * 128, sc, sh, ATOT, 126, 1);
  }

  // ---- stream fuse -> anchor_feat; gi = [anchor_feat(253) | bn(anchor_pos)(3)] ----
  gemm(strin, 256, wstr, 256, F(41), stro, 253, ATOT, 253, 256, 0);
  bn_stats(stro, 253, 0, ATOT, 253, F(42), F(43));                                // bn_fuse
  k_bn_apply32<<<dim3(512), dim3(256), 0, stream>>>(stro, 253, 0, gi32, 256, 0, sc, sh, ATOT, 253, 1);
  k_gather_pos<<<dim3(8), dim3(256), 0, stream>>>(x, aidx, apos);
  bn_stats(apos, 3, 0, ATOT, 3, F(6), F(7));                                      // pos_bn
  k_bn_apply32<<<dim3(64), dim3(256), 0, stream>>>(apos, 3, 0, gi32, 256, 253, sc, sh, ATOT, 3, 0);
  k_cvt16<<<dim3(512), dim3(256), 0, stream>>>(gi32, 256, 0, gi16, 256, 0, ATOT, 256, 0);

  // ---- transformer ----
  gemm(gi16, 256, wai, 256, F(45), qkv, 768, ATOT, 768, 256, 0);                  // qkv
  k_attention<<<dim3(64), dim3(128), 0, stream>>>(qkv, o32);
  k_cvt16<<<dim3(512), dim3(256), 0, stream>>>(o32, 256, 0, o16, 256, 0, ATOT, 256, 0);
  gemm(o16, 256, wao, 256, F(47), tA32, 256, ATOT, 256, 256, 0);                  // attn_out
  k_layernorm<<<dim3(ATOT), dim3(256), 0, stream>>>(tA32, gi32, F(48), F(49), h132);
  k_cvt16<<<dim3(512), dim3(256), 0, stream>>>(h132, 256, 0, h116, 256, 0, ATOT, 256, 0);
  gemm(h116, 256, wf1, 256, F(53), ff132, 506, ATOT, 506, 256, 1);                // ff1 + relu
  k_cvt16<<<dim3(1024), dim3(256), 0, stream>>>(ff132, 506, 0, ff116, 512, 0, ATOT, 506, 0);
  gemm(ff116, 512, wf2, 512, F(55), tA32, 256, ATOT, 256, 512, 0);                // ff2
  k_layernorm<<<dim3(ATOT), dim3(256), 0, stream>>>(tA32, h132, F(50), F(51), ag32);

  // ---- KNN interp + classifier input build ----
  hipMemsetAsync(clsin, 0, (size_t)NPTS * 288 * 2, stream);
  k_knn<<<dim3(NPTS), dim3(128), 0, stream>>>(x, apos, ag32, xn, clsin);

  // ---- classifier ----
  gemm(clsin, 288, wc1, 288, F(57), c132, 64, NPTS, 64, 288, 1);                  // cls1 + relu
  bn_stats(c132, 64, 0, NPTS, 64, F(58), F(59));                                  // cbn1
  k_bn_apply16<<<dim3(4096), dim3(256), 0, stream>>>(c132, 64, 0, c116, 64, 0, sc, sh, NPTS, 64, 0);
  gemm(c116, 64, wc2, 64, F(61), c232, 32, NPTS, 32, 64, 1);                      // cls2 + relu
  bn_stats(c232, 32, 0, NPTS, 32, F(62), F(63));                                  // cbn2
  k_bn_apply16<<<dim3(4096), dim3(256), 0, stream>>>(c232, 32, 0, c216, 32, 0, sc, sh, NPTS, 32, 0);
  gemm(c216, 32, wc3, 32, F(65), (float*)d_out, 1, NPTS, 1, 32, 0);               // cls3 -> out
}
